// MambaBlock_37134287241961
// MI455X (gfx1250) — matrix operand order repaired and checked by reading, not compiled, not run on hardware
//
#include <hip/hip_runtime.h>
#include <hip/hip_bf16.h>

// ---------------------------------------------------------------------------
// Types for CDNA5 WMMA (wave32)
// ---------------------------------------------------------------------------
typedef __attribute__((ext_vector_type(16))) __bf16 v16bf;
typedef __attribute__((ext_vector_type(8)))  __bf16 v8bf;
typedef __attribute__((ext_vector_type(8)))  float  v8f;

__device__ __forceinline__ __bf16 tobf(float f) { return (__bf16)f; }

// Problem constants
#define BB    8
#define CC    256
#define HWL   1024          // H*W == L
#define BL    8192          // BB * HWL
#define DIN   512           // d_inner
#define DST   16            // d_state
#define DTR   16            // dt_rank
#define NDBL  48            // dt_rank + 2*d_state
#define HID   1024          // mlp hidden

// ---------------------------------------------------------------------------
// Weight transpose + convert: W (K x N, f32 row-major) -> WT (N x K, bf16)
// ---------------------------------------------------------------------------
__global__ void k_wt_bf16(const float* __restrict__ W, __bf16* __restrict__ WT,
                          int K, int N) {
    int idx = blockIdx.x * blockDim.x + threadIdx.x;
    if (idx >= K * N) return;
    int n = idx % N;
    int k = idx / N;
    WT[(size_t)n * K + k] = tobf(W[idx]);
}

// ---------------------------------------------------------------------------
// LN1 fused with NCHW -> (B,L,C) transpose.  One block per row (b,l).
// ---------------------------------------------------------------------------
__global__ __launch_bounds__(256)
void k_ln1_transpose(const float* __restrict__ x, const float* __restrict__ g,
                     const float* __restrict__ be, float* __restrict__ xt,
                     __bf16* __restrict__ xn) {
    int row = blockIdx.x;                 // b*HWL + l
    int b = row >> 10, l = row & (HWL - 1);
    int c = threadIdx.x;                  // 0..255
    float v = x[(((size_t)b * CC + c) << 10) + l];

    __shared__ float red[256];
    red[c] = v; __syncthreads();
    for (int s = 128; s > 0; s >>= 1) { if (c < s) red[c] += red[c + s]; __syncthreads(); }
    float mu = red[0] * (1.f / CC);
    __syncthreads();
    float dv = v - mu;
    red[c] = dv * dv; __syncthreads();
    for (int s = 128; s > 0; s >>= 1) { if (c < s) red[c] += red[c + s]; __syncthreads(); }
    float var = red[0] * (1.f / CC);

    float xnv = dv * rsqrtf(var + 1e-5f) * g[c] + be[c];
    xt[(size_t)row * CC + c] = v;
    xn[(size_t)row * CC + c] = tobf(xnv);
}

// ---------------------------------------------------------------------------
// LN2: row-major input, bf16 output only (residual x2 stays in place).
// ---------------------------------------------------------------------------
__global__ __launch_bounds__(256)
void k_ln2(const float* __restrict__ xin, const float* __restrict__ g,
           const float* __restrict__ be, __bf16* __restrict__ xn) {
    int row = blockIdx.x;
    int c = threadIdx.x;
    float v = xin[(size_t)row * CC + c];
    __shared__ float red[256];
    red[c] = v; __syncthreads();
    for (int s = 128; s > 0; s >>= 1) { if (c < s) red[c] += red[c + s]; __syncthreads(); }
    float mu = red[0] * (1.f / CC);
    __syncthreads();
    float dv = v - mu;
    red[c] = dv * dv; __syncthreads();
    for (int s = 128; s > 0; s >>= 1) { if (c < s) red[c] += red[c + s]; __syncthreads(); }
    float var = red[0] * (1.f / CC);
    xn[(size_t)row * CC + c] = tobf(dv * rsqrtf(var + 1e-5f) * g[c] + be[c]);
}

// ---------------------------------------------------------------------------
// Depthwise causal conv(4) + SiLU on the xm half of xz; also silu(z).
// ---------------------------------------------------------------------------
__global__ __launch_bounds__(256)
void k_conv_silu(const float* __restrict__ xz, const float* __restrict__ cw,
                 const float* __restrict__ cb, float* __restrict__ xmc,
                 __bf16* __restrict__ xmcb, float* __restrict__ sz) {
    int idx = blockIdx.x * blockDim.x + threadIdx.x;   // BL*DIN
    if (idx >= BL * DIN) return;
    int d  = idx & (DIN - 1);
    int bl = idx >> 9;              // row = b*L + l
    int l  = bl & (HWL - 1);

    float acc = cb[d];
#pragma unroll
    for (int k = 0; k < 4; ++k) {
        int ls = l - 3 + k;
        if (ls >= 0)
            acc = fmaf(xz[((size_t)(bl - 3 + k)) * (2 * DIN) + d], cw[d * 4 + k], acc);
    }
    float sv = acc / (1.f + __expf(-acc));             // silu
    xmc[idx]  = sv;
    xmcb[idx] = tobf(sv);

    float zv = xz[(size_t)bl * (2 * DIN) + DIN + d];
    sz[idx] = zv / (1.f + __expf(-zv));
}

// ---------------------------------------------------------------------------
// dt projection: dt = softplus(dbl[:, :16] @ W_dt + b_dt). K=16 -> plain VALU.
// ---------------------------------------------------------------------------
__global__ __launch_bounds__(256)
void k_dtproj(const float* __restrict__ dbl, const float* __restrict__ Wdt,
              const float* __restrict__ bdt, float* __restrict__ dt) {
    int idx = blockIdx.x * blockDim.x + threadIdx.x;   // BL*DIN
    if (idx >= BL * DIN) return;
    int d   = idx & (DIN - 1);
    int row = idx >> 9;
    const float* dr = dbl + (size_t)row * NDBL;
    float acc = bdt[d];
#pragma unroll
    for (int r = 0; r < DTR; ++r)
        acc = fmaf(dr[r], Wdt[r * DIN + d], acc);
    dt[idx] = (acc > 20.f) ? acc : log1pf(__expf(acc));
}

// ---------------------------------------------------------------------------
// Selective scan. One thread per (b, d): 16 states in registers, loop L.
// B/C rows are wave-uniform -> readfirstlane so they lower to scalar loads.
// ---------------------------------------------------------------------------
__global__ __launch_bounds__(256)
void k_scan(const float* __restrict__ dt, const float* __restrict__ xmc,
            const float* __restrict__ dbl, const float* __restrict__ sz,
            const float* __restrict__ Alog, const float* __restrict__ Dp,
            __bf16* __restrict__ yb) {
    int t = blockIdx.x * blockDim.x + threadIdx.x;     // BB*DIN = 4096
    if (t >= BB * DIN) return;
    int d = t & (DIN - 1);
    int b = t >> 9;

    float a[DST];
#pragma unroll
    for (int s = 0; s < DST; ++s) a[s] = -__expf(Alog[d * DST + s]);
    float Dd = Dp[d];

    float h[DST];
#pragma unroll
    for (int s = 0; s < DST; ++s) h[s] = 0.f;

    int rbase = b * HWL;
    for (int l = 0; l < HWL; ++l) {
        int row = rbase + l;
        float dtv = dt[(size_t)row * DIN + d];
        float xv  = xmc[(size_t)row * DIN + d];
        // row is uniform across the wave (each block spans d of one (b,l) stream)
        int urow = __builtin_amdgcn_readfirstlane(row);
        const float* dr = dbl + (size_t)urow * NDBL;

        float Bv[DST], Cv[DST];
#pragma unroll
        for (int q = 0; q < 4; ++q) {
            float4 tb = *(const float4*)(dr + DTR + 4 * q);
            float4 tc = *(const float4*)(dr + DTR + DST + 4 * q);
            Bv[4 * q + 0] = tb.x; Bv[4 * q + 1] = tb.y; Bv[4 * q + 2] = tb.z; Bv[4 * q + 3] = tb.w;
            Cv[4 * q + 0] = tc.x; Cv[4 * q + 1] = tc.y; Cv[4 * q + 2] = tc.z; Cv[4 * q + 3] = tc.w;
        }

        float y = 0.f;
        float dx = dtv * xv;
#pragma unroll
        for (int s = 0; s < DST; ++s) {
            float e = __expf(dtv * a[s]);
            h[s] = fmaf(e, h[s], dx * Bv[s]);
            y = fmaf(h[s], Cv[s], y);
        }
        float yv = fmaf(xv, Dd, y);
        yb[(size_t)row * DIN + d] = tobf(yv * sz[(size_t)row * DIN + d]);
    }
}

// ---------------------------------------------------------------------------
// bf16 WMMA GEMM, fully static shapes, software-pipelined:
//   C(M x NLEN, f32) = A(M x KLEN, bf16 row-major) x BT(NLEN x KLEN, bf16)^T
// One wave -> 16(M) x NT*16(N) tile.  Fragments for step k+32 are loaded
// BEFORE the WMMAs of step k consume the current fragments, so the global
// load latency is hidden behind NT back-to-back v_wmma ops.
// ---------------------------------------------------------------------------
#define EP_STORE  0   // outF = acc
#define EP_RES    1   // outF = acc + res
#define EP_GELU   2   // outB = bf16(gelu(acc + bias))
#define EP_FINAL  3   // d_out[(b,C,l)] = acc + bias + res   (NCHW transpose)

union Frag { v16bf v; v8bf h[2]; };

template <int MODE, int NLEN, int KLEN, int NT>
__global__ __launch_bounds__(128)
void k_gemm_bf16(const __bf16* __restrict__ A, const __bf16* __restrict__ BT,
                 const float* __restrict__ bias, const float* __restrict__ res,
                 float* __restrict__ outF, __bf16* __restrict__ outB) {
    const int wave = threadIdx.x >> 5;
    const int lane = threadIdx.x & 31;
    const int half = lane >> 4;
    const int mr   = lane & 15;          // A row-in-tile, also B col-in-tile

    const int m0 = (blockIdx.y * 4 + wave) * 16;
    const int n0 = blockIdx.x * (NT * 16);

    v8f acc[NT];
#pragma unroll
    for (int t = 0; t < NT; ++t) acc[t] = (v8f)(0.f);

    const __bf16* arow = A + (size_t)(m0 + mr) * KLEN + half * 8;
    const __bf16* brow[NT];
#pragma unroll
    for (int t = 0; t < NT; ++t)
        brow[t] = BT + (size_t)(n0 + t * 16 + mr) * KLEN + half * 8;

    // ---- pipeline prologue: load fragments for k = 0 ----
    Frag fa, fb[NT];
    fa.h[0] = *(const v8bf*)(arow);
    fa.h[1] = *(const v8bf*)(arow + 16);
#pragma unroll
    for (int t = 0; t < NT; ++t) {
        fb[t].h[0] = *(const v8bf*)(brow[t]);
        fb[t].h[1] = *(const v8bf*)(brow[t] + 16);
    }

#pragma unroll
    for (int k = 0; k < KLEN; k += 32) {
        // ---- issue loads for the NEXT step before consuming current ----
        Frag na, nb[NT];
        if (k + 32 < KLEN) {
            na.h[0] = *(const v8bf*)(arow + k + 32);
            na.h[1] = *(const v8bf*)(arow + k + 48);
#pragma unroll
            for (int t = 0; t < NT; ++t) {
                nb[t].h[0] = *(const v8bf*)(brow[t] + k + 32);
                nb[t].h[1] = *(const v8bf*)(brow[t] + k + 48);
            }
        }
        if ((k & 127) == 0 && k + 256 < KLEN)
            __builtin_prefetch(arow + k + 256, 0, 0);   // global_prefetch_b8

        // ---- NT back-to-back WMMAs on distinct accumulators ----
#pragma unroll
        for (int t = 0; t < NT; ++t)
            acc[t] = __builtin_amdgcn_wmma_f32_16x16x32_bf16(
                false, fa.v, false, fb[t].v, (short)0, acc[t], false, false);

        // ---- rotate (removed by SSA renaming under full unroll) ----
        fa = na;
#pragma unroll
        for (int t = 0; t < NT; ++t) fb[t] = nb[t];
    }

#pragma unroll
    for (int t = 0; t < NT; ++t) {
        int col = n0 + t * 16 + mr;
#pragma unroll
        for (int r = 0; r < 8; ++r) {
            int row = m0 + r + half * 8;
            float v = acc[t][r];
            if (MODE == EP_STORE) {
                outF[(size_t)row * NLEN + col] = v;
            } else if (MODE == EP_RES) {
                outF[(size_t)row * NLEN + col] = v + res[(size_t)row * NLEN + col];
            } else if (MODE == EP_GELU) {
                float u = v + bias[col];
                float gl = 0.5f * u * (1.f + erff(u * 0.70710678118654752f));
                outB[(size_t)row * NLEN + col] = tobf(gl);
            } else { // EP_FINAL: row = b*L + l ; out shape (B, C, H, W), NLEN == C
                float u = v + bias[col] + res[(size_t)row * NLEN + col];
                int b = row >> 10, l = row & (HWL - 1);
                outF[(((size_t)b * NLEN + col) << 10) + l] = u;
            }
        }
    }
}

// ---------------------------------------------------------------------------
// Launcher
// ---------------------------------------------------------------------------
extern "C" void kernel_launch(void* const* d_in, const int* in_sizes, int n_in,
                              void* d_out, int out_size, void* d_ws, size_t ws_size,
                              hipStream_t stream) {
    const float* x      = (const float*)d_in[0];
    const float* ln1_g  = (const float*)d_in[1];
    const float* ln1_b  = (const float*)d_in[2];
    const float* ln2_g  = (const float*)d_in[3];
    const float* ln2_b  = (const float*)d_in[4];
    const float* W_in   = (const float*)d_in[5];
    const float* conv_w = (const float*)d_in[6];
    const float* conv_b = (const float*)d_in[7];
    const float* W_x    = (const float*)d_in[8];
    const float* W_dt   = (const float*)d_in[9];
    const float* b_dt   = (const float*)d_in[10];
    const float* A_log  = (const float*)d_in[11];
    const float* D_par  = (const float*)d_in[12];
    const float* W_out  = (const float*)d_in[13];
    const float* W1     = (const float*)d_in[14];
    const float* b1     = (const float*)d_in[15];
    const float* W2     = (const float*)d_in[16];
    const float* b2     = (const float*)d_in[17];

    // workspace carve-out
    char* w = (char*)d_ws;
    auto carve = [&](size_t bytes) -> char* {
        char* p = w;
        w += (bytes + 255) & ~(size_t)255;
        return p;
    };
    float*  xt    = (float*) carve((size_t)BL * CC * 4);        // residual 1
    __bf16* xnb   = (__bf16*)carve((size_t)BL * CC * 2);        // LN1 out
    float*  xz    = (float*) carve((size_t)BL * 2 * DIN * 4);   // W_in GEMM out
    float*  xmc   = (float*) carve((size_t)BL * DIN * 4);       // conv+silu f32
    __bf16* xmcb  = (__bf16*)carve((size_t)BL * DIN * 2);       // conv+silu bf16
    float*  szb   = (float*) carve((size_t)BL * DIN * 4);       // silu(z)
    float*  dblb  = (float*) carve((size_t)BL * NDBL * 4);      // W_x GEMM out
    float*  dtb   = (float*) carve((size_t)BL * DIN * 4);       // softplus dt
    __bf16* yb    = (__bf16*)carve((size_t)BL * DIN * 2);       // scan out
    float*  x2    = (float*) carve((size_t)BL * CC * 4);        // residual 2
    __bf16* xn2b  = (__bf16*)carve((size_t)BL * CC * 2);        // LN2 out
    __bf16* hb    = (__bf16*)carve((size_t)BL * HID * 2);       // gelu(mlp1)
    __bf16* WinT  = (__bf16*)carve((size_t)(2 * DIN) * CC * 2);
    __bf16* WxT   = (__bf16*)carve((size_t)NDBL * DIN * 2);
    __bf16* WoutT = (__bf16*)carve((size_t)CC * DIN * 2);
    __bf16* W1T   = (__bf16*)carve((size_t)HID * CC * 2);
    __bf16* W2T   = (__bf16*)carve((size_t)CC * HID * 2);

    // --- weight transposes -> bf16 (N x K layouts) ---
    {
        int n;
        n = CC * 2 * DIN; k_wt_bf16<<<(n + 255) / 256, 256, 0, stream>>>(W_in,  WinT,  CC,  2 * DIN);
        n = DIN * NDBL;   k_wt_bf16<<<(n + 255) / 256, 256, 0, stream>>>(W_x,   WxT,   DIN, NDBL);
        n = DIN * CC;     k_wt_bf16<<<(n + 255) / 256, 256, 0, stream>>>(W_out, WoutT, DIN, CC);
        n = CC * HID;     k_wt_bf16<<<(n + 255) / 256, 256, 0, stream>>>(W1,    W1T,   CC,  HID);
        n = HID * CC;     k_wt_bf16<<<(n + 255) / 256, 256, 0, stream>>>(W2,    W2T,   HID, CC);
    }

    // --- stage 1: transpose + LN1 ---
    k_ln1_transpose<<<BL, 256, 0, stream>>>(x, ln1_g, ln1_b, xt, xnb);

    // --- stage 2: xz = xn @ W_in   (M=8192, K=256, N=1024) ---
    k_gemm_bf16<EP_STORE, 2 * DIN, CC, 4><<<dim3((2 * DIN) / 64, BL / 64), 128, 0, stream>>>(
        xnb, WinT, nullptr, nullptr, xz, nullptr);

    // --- stage 3: depthwise conv + silu, silu(z) ---
    k_conv_silu<<<(BL * DIN + 255) / 256, 256, 0, stream>>>(xz, conv_w, conv_b,
                                                            xmc, xmcb, szb);

    // --- stage 4: dbl = xm @ W_x   (M=8192, K=512, N=48, NT=3) ---
    k_gemm_bf16<EP_STORE, NDBL, DIN, 3><<<dim3(1, BL / 64), 128, 0, stream>>>(
        xmcb, WxT, nullptr, nullptr, dblb, nullptr);

    // --- stage 5: dt = softplus(dbl[:, :16] @ W_dt + b_dt) ---
    k_dtproj<<<(BL * DIN + 255) / 256, 256, 0, stream>>>(dblb, W_dt, b_dt, dtb);

    // --- stage 6: selective scan (fused *D and *silu(z)) ---
    k_scan<<<(BB * DIN + 255) / 256, 256, 0, stream>>>(dtb, xmc, dblb, szb,
                                                       A_log, D_par, yb);

    // --- stage 7: x2 = y @ W_out + res   (M=8192, K=512, N=256) ---
    k_gemm_bf16<EP_RES, CC, DIN, 4><<<dim3(CC / 64, BL / 64), 128, 0, stream>>>(
        yb, WoutT, nullptr, xt, x2, nullptr);

    // --- stage 8: LN2 ---
    k_ln2<<<BL, 256, 0, stream>>>(x2, ln2_g, ln2_b, xn2b);

    // --- stage 9: h = gelu(xn2 @ W1 + b1)  (M=8192, K=256, N=1024) ---
    k_gemm_bf16<EP_GELU, HID, CC, 4><<<dim3(HID / 64, BL / 64), 128, 0, stream>>>(
        xn2b, W1T, b1, nullptr, nullptr, hb);

    // --- stage 10: out = h @ W2 + b2 + x2, stored NCHW-transposed ---
    k_gemm_bf16<EP_FINAL, CC, HID, 4><<<dim3(CC / 64, BL / 64), 128, 0, stream>>>(
        hb, W2T, b2, x2, (float*)d_out, nullptr);
}
